// MGFRTE_38482906972893
// MI455X (gfx1250) — matrix operand order repaired and checked by reading, not compiled, not run on hardware
//
#include <hip/hip_runtime.h>
#include <hip/hip_bf16.h>

typedef __attribute__((ext_vector_type(16))) __bf16 v16bf;
typedef __attribute__((ext_vector_type(8)))  float  v8f;

union Frag16 { v16bf v; uint4 q[2]; };

// Problem constants (reference setup_inputs)
constexpr int B_ = 2, N_ = 5, K_ = 5, Q_ = 4;
constexpr int Ls = 128, Lq = 128, D = 768, T = 5;
constexpr int BNK = B_ * N_ * K_;      // 50
constexpr int BNQ = B_ * N_ * Q_;      // 40
constexpr int BQ  = B_ * N_ * Q_ * N_; // 200
constexpr int K4  = 4 * D;             // 3072 (fused dim)
constexpr int APITCH = 40;             // bf16 elems per A-tile row (80B, 16B aligned, bank-tiling-free)

// ---------------------------------------------------------------------------
// Kernel 1: per-(b,n,k) tag prototypes.  sums[t][d] = sum_l w[l]*emb[l][d] for
// argmax-label t; tag 0 weight gated by support_mask. One block per bnk.
// ---------------------------------------------------------------------------
__global__ __launch_bounds__(256) void k_proto(
    const float* __restrict__ emb,   // [BNK][Ls][D]
    const float* __restrict__ smask, // [BNK][Ls]
    const float* __restrict__ sel,   // [BNK][Ls][T]
    float* __restrict__ protob)      // [BNK][T][D]
{
  __shared__ int   labL[Ls];
  __shared__ float wL[Ls];
  __shared__ float cntL[T];
  const int bnk = blockIdx.x;
  const int tid = threadIdx.x;

  if (tid < Ls) {
    const float* lr = sel + ((size_t)bnk * Ls + tid) * T;
    int best = 0; float bv = lr[0];
    #pragma unroll
    for (int t = 1; t < T; ++t) { float v = lr[t]; if (v > bv) { bv = v; best = t; } }
    labL[tid] = best;
    wL[tid] = (best == 0) ? smask[(size_t)bnk * Ls + tid] : 1.0f;
  }
  __syncthreads();
  if (tid < T) {
    float c = 0.f;
    for (int l = 0; l < Ls; ++l) if (labL[l] == tid) c += wL[l];
    cntL[tid] = c;
  }
  __syncthreads();

  const float* er = emb + (size_t)bnk * Ls * D;
  for (int d = tid; d < D; d += 256) {
    float a0 = 0, a1 = 0, a2 = 0, a3 = 0, a4 = 0;
    for (int l = 0; l < Ls; ++l) {
      float v = wL[l] * er[(size_t)l * D + d];
      int t = labL[l];
      a0 += (t == 0) ? v : 0.f;  a1 += (t == 1) ? v : 0.f;
      a2 += (t == 2) ? v : 0.f;  a3 += (t == 3) ? v : 0.f;
      a4 += (t == 4) ? v : 0.f;
    }
    float acc[T] = {a0, a1, a2, a3, a4};
    float* o = protob + (size_t)bnk * T * D + d;
    #pragma unroll
    for (int t = 0; t < T; ++t) {
      float c = cntL[t];
      o[(size_t)t * D] = (c > 0.f) ? acc[t] / fmaxf(c, 1.0f) : 0.f;
    }
  }
}

// ---------------------------------------------------------------------------
// Kernel 2: mean over the K shots -> proto [B*N][T][D]
// ---------------------------------------------------------------------------
__global__ __launch_bounds__(256) void k_proto_mean(
    const float* __restrict__ protob, float* __restrict__ proto)
{
  int i = blockIdx.x * blockDim.x + threadIdx.x;
  if (i >= B_ * N_ * T * D) return;
  int td = i % (T * D);
  int bn = i / (T * D);
  float s = 0.f;
  #pragma unroll
  for (int k = 0; k < K_; ++k) s += protob[(size_t)(bn * K_ + k) * T * D + td];
  proto[i] = s * (1.0f / K_);
}

// ---------------------------------------------------------------------------
// Kernel 3: fp32 -> bf16 weight conversion (row-major [768][3072] is already
// the K-contiguous-per-output-column layout the B fragments want).
// ---------------------------------------------------------------------------
__global__ __launch_bounds__(256) void k_cvt_bf16(
    const float* __restrict__ src, __bf16* __restrict__ dst, int n)
{
  int i = blockIdx.x * blockDim.x + threadIdx.x;
  if (i < n) dst[i] = (__bf16)src[i];
}

// ---------------------------------------------------------------------------
// Fragment loaders (contiguous k order, wave32).
// A 16x32 bf16: lane<16 -> M=lane,    K = 0..15 (contiguous)
//               lane>=16 -> M=lane-16, K = 16..31
// ---------------------------------------------------------------------------
__device__ __forceinline__ v16bf load_a_frag(const __bf16* row /* 40-elem pitch row base */,
                                             int lane)
{
  Frag16 fa;
  const int ak = (lane < 16) ? 0 : 16;
  fa.q[0] = *(const uint4*)(row + ak);
  fa.q[1] = *(const uint4*)(row + ak + 8);
  return fa.v;
}
// B 32x16 bf16: lane<16 -> N=lane,    K = 0..15 (contiguous)
//               lane>=16 -> N=lane-16, K = 16..31
__device__ __forceinline__ v16bf load_b_frag_global(const __bf16* __restrict__ wrow /* col row base + kt*32 */,
                                                    int lane)
{
  Frag16 fb;
  const __bf16* p = wrow + ((lane < 16) ? 0 : 16);
  fb.q[0] = ((const uint4*)p)[0];
  fb.q[1] = ((const uint4*)p)[1];
  return fb.v;
}

// ---------------------------------------------------------------------------
// Kernel 4: one block per (b, nq, n) pair. Attention + softmaxes + support_,
// then es (M=16, rows 0..4 valid) and eq (M=128) GEMMs with on-the-fly fused
// A tiles, bias+relu, masked max/mean pooling -> eq_agg/es_agg.
// ---------------------------------------------------------------------------
__global__ __launch_bounds__(256) void k_pair(
    const float* __restrict__ qte,    // [BNQ][Lq][D]
    const float* __restrict__ qmask,  // [BNQ][Lq]
    const float* __restrict__ proto,  // [B*N][T][D]
    const __bf16* __restrict__ wbf,   // proj_w bf16 [D][4D]
    const float* __restrict__ projb,  // [D]
    float* __restrict__ eq_agg,       // [BQ][2D]
    float* __restrict__ es_agg)       // [BQ][2D]
{
  __shared__ float protoL[T][D];
  __shared__ float suppL[T][D];
  __shared__ float P2[Lq][8];            // softmax over tags, padded
  __shared__ float qmaskL[Lq];
  __shared__ float qinvL;
  __shared__ __bf16 Atile[Lq][APITCH];
  union Scratch {
    struct { float att[T][Lq]; float P1[T][Lq]; } sm;
    struct { float rmax[8][256]; float rsum[8][256]; } red;
  };
  __shared__ Scratch scr;

  const int p    = blockIdx.x;
  const int n    = p % N_;
  const int bnq  = p / N_;
  const int b    = bnq / (N_ * Q_);
  const int tid  = threadIdx.x;
  const int lane = tid & 31;
  const int wave = tid >> 5;

  const float* qrow = qte + (size_t)bnq * Lq * D;
  const float* pr   = proto + (size_t)(b * N_ + n) * T * D;

  for (int i = tid; i < T * D; i += 256) ((float*)protoL)[i] = pr[i];
  if (tid < Lq) qmaskL[tid] = qmask[(size_t)bnq * Lq + tid];
  __syncthreads();
  if (tid == 0) { float s = 0.f; for (int q = 0; q < Lq; ++q) s += qmaskL[q]; qinvL = 1.0f / s; }

  // ---- att[s][q] = proto_s . query_q + 100*qmask[q] ----
  for (int e = tid; e < T * Lq; e += 256) {
    int s = e / Lq, q = e % Lq;
    const float* qp = qrow + (size_t)q * D;
    const float* pp = protoL[s];
    float acc = 0.f;
    for (int d = 0; d < D; ++d) acc += pp[d] * qp[d];
    scr.sm.att[s][q] = acc + 100.0f * qmaskL[q];
  }
  __syncthreads();

  // ---- P1: softmax over q (waves 0..4, one tag row each) ----
  if (wave < T) {
    const int s = wave;
    float m = -1e30f;
    for (int q = lane; q < Lq; q += 32) m = fmaxf(m, scr.sm.att[s][q]);
    #pragma unroll
    for (int o = 16; o > 0; o >>= 1) m = fmaxf(m, __shfl_xor(m, o, 32));
    float sum = 0.f;
    for (int q = lane; q < Lq; q += 32) { float e = __expf(scr.sm.att[s][q] - m); scr.sm.P1[s][q] = e; sum += e; }
    #pragma unroll
    for (int o = 16; o > 0; o >>= 1) sum += __shfl_xor(sum, o, 32);
    const float inv = 1.0f / sum;
    for (int q = lane; q < Lq; q += 32) scr.sm.P1[s][q] *= inv;
  }
  // ---- P2: softmax over the 5 tags per query token ----
  if (tid < Lq) {
    const int q = tid;
    float m = -1e30f;
    #pragma unroll
    for (int s = 0; s < T; ++s) m = fmaxf(m, scr.sm.att[s][q]);
    float ev[T]; float sum = 0.f;
    #pragma unroll
    for (int s = 0; s < T; ++s) { ev[s] = __expf(scr.sm.att[s][q] - m); sum += ev[s]; }
    const float inv = 1.0f / sum;
    #pragma unroll
    for (int s = 0; s < T; ++s) P2[q][s] = ev[s] * inv;
  }
  __syncthreads();

  // ---- support_ = P1 @ query : [T][D] into LDS ----
  {
    int sArr[15], dArr[15];
    float acc[15];
    #pragma unroll
    for (int j = 0; j < 15; ++j) { int e = tid + 256 * j; sArr[j] = e / D; dArr[j] = e % D; acc[j] = 0.f; }
    for (int q = 0; q < Lq; ++q) {
      const float* qp = qrow + (size_t)q * D;
      #pragma unroll
      for (int j = 0; j < 15; ++j) acc[j] += scr.sm.P1[sArr[j]][q] * qp[dArr[j]];
    }
    #pragma unroll
    for (int j = 0; j < 15; ++j) suppL[sArr[j]][dArr[j]] = acc[j];
  }
  __syncthreads();

  // =========================== es GEMM (M=16) ===========================
  v8f eacc[6];
  #pragma unroll
  for (int i = 0; i < 6; ++i)
    #pragma unroll
    for (int j = 0; j < 8; ++j) eacc[i][j] = 0.f;

  for (int kt = 0; kt < 96; ++kt) {
    const int seg = kt / 24;
    const int kk  = (kt % 24) * 32;
    for (int e = tid; e < 16 * 32; e += 256) {
      int r = e >> 5, c = e & 31, d = kk + c;
      float v = 0.f;
      if (r < T) {
        float m1 = protoL[r][d];
        float m2 = suppL[r][d];
        v = (seg == 0) ? m1 : (seg == 1) ? m2 : (seg == 2) ? fabsf(m1 - m2) : m1 * m2;
      }
      Atile[r][c] = (__bf16)v;
    }
    __syncthreads();
    const v16bf fa = load_a_frag(&Atile[lane & 15][0], lane);
    #pragma unroll
    for (int i = 0; i < 6; ++i) {
      const int ncol = (wave * 6 + i) * 16 + (lane & 15);
      const v16bf fb = load_b_frag_global(wbf + (size_t)ncol * K4 + kt * 32, lane);
      eacc[i] = __builtin_amdgcn_wmma_f32_16x16x32_bf16(false, fa, false, fb,
                                                        (short)0, eacc[i], false, false);
    }
    __syncthreads();
  }
  // epilogue: relu+bias, masked max/mean over the 5 valid tag rows
  if (lane < 16) {
    #pragma unroll
    for (int i = 0; i < 6; ++i) {
      const int ncol = (wave * 6 + i) * 16 + lane;
      const float bias = projb[ncol];
      float mx = -1e30f, sm = 0.f;
      #pragma unroll
      for (int v = 0; v < T; ++v) {           // rows 0..4 live in vgprs 0..4, lanes<16
        float h = fmaxf(eacc[i][v] + bias, 0.f);
        mx = fmaxf(mx, h); sm += h;
      }
      es_agg[(size_t)p * (2 * D) + ncol]     = mx;
      es_agg[(size_t)p * (2 * D) + D + ncol] = sm * (1.0f / T);
    }
  }
  __syncthreads();

  // =========================== eq GEMM (M=128) ===========================
  for (int npass = 0; npass < 3; ++npass) {
    const int cbase = npass * 256;
    v8f acc[16];
    #pragma unroll
    for (int i = 0; i < 16; ++i)
      #pragma unroll
      for (int j = 0; j < 8; ++j) acc[i][j] = 0.f;

    for (int kt = 0; kt < 96; ++kt) {
      const int seg = kt / 24;
      const int kk  = (kt % 24) * 32;
      {
        const int c = tid & 31, q0 = tid >> 5, d = kk + c;
        #pragma unroll
        for (int i = 0; i < 16; ++i) {
          const int q = q0 + i * 8;
          const float m1 = qrow[(size_t)q * D + d];
          float val;
          if (seg == 0) val = m1;
          else {
            const float m2 = qmaskL[q] *
              (P2[q][0] * protoL[0][d] + P2[q][1] * protoL[1][d] + P2[q][2] * protoL[2][d] +
               P2[q][3] * protoL[3][d] + P2[q][4] * protoL[4][d]);
            val = (seg == 1) ? m2 : (seg == 2) ? fabsf(m1 - m2) : m1 * m2;
          }
          Atile[q][c] = (__bf16)val;
        }
      }
      __syncthreads();
      const v16bf fa = load_a_frag(&Atile[wave * 16 + (lane & 15)][0], lane);
      #pragma unroll
      for (int i = 0; i < 16; ++i) {
        const int ncol = cbase + i * 16 + (lane & 15);
        const v16bf fb = load_b_frag_global(wbf + (size_t)ncol * K4 + kt * 32, lane);
        acc[i] = __builtin_amdgcn_wmma_f32_16x16x32_bf16(false, fa, false, fb,
                                                         (short)0, acc[i], false, false);
      }
      __syncthreads();
    }
    // epilogue: relu+bias, unmasked max & sum over 128 rows, cross-wave reduce
    #pragma unroll
    for (int i = 0; i < 16; ++i) {
      const int col  = cbase + i * 16 + (lane & 15);
      const float bias = projb[col];
      float mx = -1e30f, sm = 0.f;
      #pragma unroll
      for (int v = 0; v < 8; ++v) {
        float h = fmaxf(acc[i][v] + bias, 0.f);
        mx = fmaxf(mx, h); sm += h;
      }
      mx = fmaxf(mx, __shfl_xor(mx, 16, 32));  // join rows m..m+7 with m+8..m+15
      sm = sm + __shfl_xor(sm, 16, 32);
      if (lane < 16) { scr.red.rmax[wave][i * 16 + lane] = mx;
                       scr.red.rsum[wave][i * 16 + lane] = sm; }
    }
    __syncthreads();
    {
      const int col = tid;
      float mx = -1e30f, sm = 0.f;
      #pragma unroll
      for (int w = 0; w < 8; ++w) { mx = fmaxf(mx, scr.red.rmax[w][col]); sm += scr.red.rsum[w][col]; }
      eq_agg[(size_t)p * (2 * D) + cbase + col]     = mx;
      eq_agg[(size_t)p * (2 * D) + D + cbase + col] = sm * qinvL;
    }
    __syncthreads();
  }
}

// ---------------------------------------------------------------------------
// Kernel 5: relation MLP. 16 pair-rows per block; cat(eq_agg,es_agg)[16x3072]
// @ rel_w1^T -> relu -> dot rel_w2 -> logits (pair index == output flat index)
// ---------------------------------------------------------------------------
__global__ __launch_bounds__(256) void k_rel(
    const float* __restrict__ eq_agg, const float* __restrict__ es_agg,
    const __bf16* __restrict__ w1bf, const float* __restrict__ rb1,
    const float* __restrict__ w2, const float* __restrict__ rb2,
    float* __restrict__ out)
{
  __shared__ __bf16 Atile[16][APITCH];
  __shared__ float partL[8][16];
  const int blk  = blockIdx.x;
  const int tid  = threadIdx.x;
  const int lane = tid & 31;
  const int wave = tid >> 5;

  v8f acc[6];
  #pragma unroll
  for (int i = 0; i < 6; ++i)
    #pragma unroll
    for (int j = 0; j < 8; ++j) acc[i][j] = 0.f;

  for (int kt = 0; kt < 96; ++kt) {
    for (int e = tid; e < 16 * 32; e += 256) {
      int r = e >> 5, c = e & 31;
      int pp = blk * 16 + r;
      int k  = kt * 32 + c;
      float v = 0.f;
      if (pp < BQ)
        v = (k < 2 * D) ? eq_agg[(size_t)pp * (2 * D) + k]
                        : es_agg[(size_t)pp * (2 * D) + (k - 2 * D)];
      Atile[r][c] = (__bf16)v;
    }
    __syncthreads();
    const v16bf fa = load_a_frag(&Atile[lane & 15][0], lane);
    #pragma unroll
    for (int i = 0; i < 6; ++i) {
      const int ncol = (wave * 6 + i) * 16 + (lane & 15);
      const v16bf fb = load_b_frag_global(w1bf + (size_t)ncol * K4 + kt * 32, lane);
      acc[i] = __builtin_amdgcn_wmma_f32_16x16x32_bf16(false, fa, false, fb,
                                                       (short)0, acc[i], false, false);
    }
    __syncthreads();
  }

  float part[8];
  #pragma unroll
  for (int v = 0; v < 8; ++v) part[v] = 0.f;
  #pragma unroll
  for (int i = 0; i < 6; ++i) {
    const int col = (wave * 6 + i) * 16 + (lane & 15);
    const float b1 = rb1[col];
    const float wv = w2[col];
    #pragma unroll
    for (int v = 0; v < 8; ++v) part[v] += fmaxf(acc[i][v] + b1, 0.f) * wv;
  }
  #pragma unroll
  for (int v = 0; v < 8; ++v) {
    float x = part[v];
    x += __shfl_xor(x, 1, 32); x += __shfl_xor(x, 2, 32);
    x += __shfl_xor(x, 4, 32); x += __shfl_xor(x, 8, 32);
    if ((lane & 15) == 0) partL[wave][((lane >> 4) << 3) + v] = x;
  }
  __syncthreads();
  if (tid < 16) {
    float s = 0.f;
    #pragma unroll
    for (int w = 0; w < 8; ++w) s += partL[w][tid];
    const int pp = blk * 16 + tid;
    if (pp < BQ) out[pp] = s + rb2[0];
  }
}

// ---------------------------------------------------------------------------
extern "C" void kernel_launch(void* const* d_in, const int* in_sizes, int n_in,
                              void* d_out, int out_size, void* d_ws, size_t ws_size,
                              hipStream_t stream) {
  const float* d_ste   = (const float*)d_in[0];   // [50][128][768]
  const float* d_qte   = (const float*)d_in[1];   // [40][128][768]
  const float* d_smask = (const float*)d_in[2];   // [50][128]
  const float* d_qmask = (const float*)d_in[3];   // [40][128]
  const float* d_sel   = (const float*)d_in[4];   // [50][128][5]
  const float* d_projw = (const float*)d_in[5];   // [768][3072]
  const float* d_projb = (const float*)d_in[6];   // [768]
  const float* d_rw1   = (const float*)d_in[7];   // [768][3072]
  const float* d_rb1   = (const float*)d_in[8];   // [768]
  const float* d_rw2   = (const float*)d_in[9];   // [1][768]
  const float* d_rb2   = (const float*)d_in[10];  // [1]
  float* out = (float*)d_out;                     // [40][5] == [BQ]

  char* ws = (char*)d_ws;
  auto carve = [&](size_t bytes) {
    char* q = ws;
    ws += (bytes + 255) & ~(size_t)255;
    return (void*)q;
  };
  float*  protob = (float*)carve((size_t)BNK * T * D * 4);
  float*  proto  = (float*)carve((size_t)B_ * N_ * T * D * 4);
  __bf16* wbf    = (__bf16*)carve((size_t)D * K4 * 2);
  __bf16* w1bf   = (__bf16*)carve((size_t)D * K4 * 2);
  float*  eqagg  = (float*)carve((size_t)BQ * 2 * D * 4);
  float*  esagg  = (float*)carve((size_t)BQ * 2 * D * 4);

  k_proto<<<BNK, 256, 0, stream>>>(d_ste, d_smask, d_sel, protob);
  k_proto_mean<<<(B_ * N_ * T * D + 255) / 256, 256, 0, stream>>>(protob, proto);
  const int nw = D * K4;
  k_cvt_bf16<<<(nw + 255) / 256, 256, 0, stream>>>(d_projw, wbf, nw);
  k_cvt_bf16<<<(nw + 255) / 256, 256, 0, stream>>>(d_rw1, w1bf, nw);
  k_pair<<<BQ, 256, 0, stream>>>(d_qte, d_qmask, proto, wbf, d_projb, eqagg, esagg);
  k_rel<<<(BQ + 15) / 16, 256, 0, stream>>>(eqagg, esagg, w1bf, d_rb1, d_rw2, d_rb2, out);
}
